// LinearAttention_47897475285213
// MI455X (gfx1250) — matrix operand order repaired and checked by reading, not compiled, not run on hardware
//
#include <hip/hip_runtime.h>

// Problem constants (match reference)
#define Bb 8
#define Ss 4096
#define Ff 256
#define Ii 512
#define Dd 4
#define Kk 7
#define Vv 256
#define I3 (3 * Ii)
#define BSr (Bb * Ss)   // flattened rows = B*S

// chunked-scan config
#define NCHUNK 32
#define CHLEN (Ss / NCHUNK)   // 128

typedef __bf16 bf16_t;
typedef __attribute__((ext_vector_type(16))) __bf16 v16bf;
typedef __attribute__((ext_vector_type(8)))  __bf16 v8bf;
typedef __attribute__((ext_vector_type(8)))  float  v8f;

// ---------------------------------------------------------------------------
// helpers
// ---------------------------------------------------------------------------
__device__ __forceinline__ float mishf(float x) {
  float sp = (x > 20.f) ? x : __logf(1.f + __expf(x));
  return x * tanhf(sp);
}

// A-matrix fragment, 16x32 bf16 (rows = seq positions, K = channels).
// Lane L (m = L&15, half = L>>4) holds K runs [half*8, half*8+8) and
// [16+half*8, 16+half*8+8) of row (rbase+m)  -> two contiguous b128 loads.
__device__ __forceinline__ v16bf frag_a(const bf16_t* __restrict__ X, int ld,
                                        int rbase, int kbase) {
  const int lane = threadIdx.x & 31;
  const int half = lane >> 4, m = lane & 15;
  const bf16_t* p = X + (size_t)(rbase + m) * ld + kbase + half * 8;
  v8bf lo = *(const v8bf*)p;
  v8bf hi = *(const v8bf*)(p + 16);
  v16bf r;
#pragma unroll
  for (int i = 0; i < 8; ++i) { r[i] = lo[i]; r[i + 8] = hi[i]; }
  return r;
}

// Same, with a causal row shift (shift in [-6,0]); rows before the start of a
// sequence read zeros. Guard only affects the VMEM loads; lanes reconverge
// before the WMMA (EXEC all-1s requirement).
__device__ __forceinline__ v16bf frag_a_shift(const bf16_t* __restrict__ X, int ld,
                                              int rbase, int shift, int kbase) {
  const int lane = threadIdx.x & 31;
  const int half = lane >> 4, m = lane & 15;
  const int row = rbase + m;
  v16bf r;
  if (((row & (Ss - 1)) + shift) >= 0) {
    const bf16_t* p = X + (size_t)(row + shift) * ld + kbase + half * 8;
    v8bf lo = *(const v8bf*)p;
    v8bf hi = *(const v8bf*)(p + 16);
#pragma unroll
    for (int i = 0; i < 8; ++i) { r[i] = lo[i]; r[i + 8] = hi[i]; }
  } else {
#pragma unroll
    for (int i = 0; i < 16; ++i) r[i] = (__bf16)0.0f;
  }
  return r;
}

// B-matrix fragment, 32x16 bf16 = W^T tile where W is [Cout][Cin] row-major.
// Lane L (n = L&15, half = L>>4) holds K runs [half*8, half*8+8) and
// [16+half*8, 16+half*8+8) of output-channel row (obase+n) -> two b128 loads.
__device__ __forceinline__ v16bf frag_b(const bf16_t* __restrict__ W, int ld,
                                        int obase, int kbase) {
  const int lane = threadIdx.x & 31;
  const int half = lane >> 4, n = lane & 15;
  const bf16_t* p = W + (size_t)(obase + n) * ld + kbase + half * 8;
  v8bf lo = *(const v8bf*)p;
  v8bf hi = *(const v8bf*)(p + 16);
  v16bf r;
#pragma unroll
  for (int i = 0; i < 8; ++i) { r[i] = lo[i]; r[i + 8] = hi[i]; }
  return r;
}

__device__ __forceinline__ v8f wmma_bf16(v16bf a, v16bf b, v8f c) {
  return __builtin_amdgcn_wmma_f32_16x16x32_bf16(false, a, false, b, (short)0, c,
                                                 false, false);
}

// D-tile epilogues (M = seq rows, N = output channels, channel-last store)
__device__ __forceinline__ void store_mish(bf16_t* __restrict__ OUT, int Cout,
                                           int rbase, int obase, v8f acc) {
  const int lane = threadIdx.x & 31;
  const int half = lane >> 4, n = lane & 15;
#pragma unroll
  for (int r = 0; r < 8; ++r) {
    int row = rbase + r + half * 8;
    OUT[(size_t)row * Cout + obase + n] = (bf16_t)mishf(acc[r]);
  }
}

__device__ __forceinline__ void store_res(bf16_t* __restrict__ OUT,
                                          const bf16_t* __restrict__ RES, int Cout,
                                          int rbase, int obase, v8f acc) {
  const int lane = threadIdx.x & 31;
  const int half = lane >> 4, n = lane & 15;
#pragma unroll
  for (int r = 0; r < 8; ++r) {
    int row = rbase + r + half * 8;
    size_t off = (size_t)row * Cout + obase + n;
    OUT[off] = (bf16_t)((float)RES[off] + acc[r]);
  }
}

__device__ __forceinline__ void store_logits(float* __restrict__ OUT,
                                             const float* __restrict__ bias,
                                             int rbase, int vbase, v8f acc) {
  const int lane = threadIdx.x & 31;
  const int half = lane >> 4, n = lane & 15;
  const int v = vbase + n;
  const float bv = bias[v];
#pragma unroll
  for (int r = 0; r < 8; ++r) {
    int row = rbase + r + half * 8;   // flattened b*S + s
    int b = row >> 12;                // / Ss
    int s = row & (Ss - 1);
    OUT[((size_t)b * Vv + v) * Ss + s] = acc[r] + bv;
  }
}

// ---------------------------------------------------------------------------
// weight prep: fp32 -> bf16 (+ transpose for the causal-conv weights)
// ---------------------------------------------------------------------------
__global__ void k_cvt_bf16(const float* __restrict__ src, bf16_t* __restrict__ dst,
                           size_t n) {
  size_t t = (size_t)blockIdx.x * blockDim.x + threadIdx.x;
  size_t stride = (size_t)gridDim.x * blockDim.x;
  for (; t < n; t += stride) dst[t] = (bf16_t)src[t];
}

// w1: fp32 [D][3I][I][K] -> bf16 [D][K][3I][I]  (tap-major, channel-contiguous)
__global__ void k_w1_prep(const float* __restrict__ src, bf16_t* __restrict__ dst) {
  size_t n = (size_t)Dd * Kk * I3 * Ii;
  size_t t = (size_t)blockIdx.x * blockDim.x + threadIdx.x;
  size_t stride = (size_t)gridDim.x * blockDim.x;
  for (; t < n; t += stride) {
    int c = (int)(t % Ii);
    size_t u = t / Ii;
    int o = (int)(u % I3);
    size_t w = u / I3;
    int k = (int)(w % Kk);
    int d = (int)(w / Kk);
    dst[t] = (bf16_t)src[(((size_t)d * I3 + o) * Ii + c) * Kk + k];
  }
}

// ---------------------------------------------------------------------------
// embedding gather: x0/x1 [B*S][F] bf16, channel-last
// ---------------------------------------------------------------------------
__global__ void k_embed(const int* __restrict__ inp, const float* __restrict__ emb,
                        bf16_t* __restrict__ x0, bf16_t* __restrict__ x1) {
  size_t t = (size_t)blockIdx.x * blockDim.x + threadIdx.x;
  if (t >= (size_t)BSr * Ff) return;
  size_t row = t / Ff;
  int c = (int)(t % Ff);
  int idx = inp[row];
  const float* e = emb + (size_t)idx * (2 * Ff);
  x0[row * Ff + c] = (bf16_t)e[c];
  x1[row * Ff + c] = (bf16_t)e[Ff + c];
}

// ---------------------------------------------------------------------------
// GEMM (2x2 register blocking): each wave computes a 32x32 output tile.
// Block = 8 waves covering 32 rows x 256 output channels.
// OUT[r][o] = mish( sum_c X[r][c] * W[o][c] )
// ---------------------------------------------------------------------------
__global__ void k_gemm_mish(const bf16_t* __restrict__ X, const bf16_t* __restrict__ W,
                            bf16_t* __restrict__ OUT, int Cin, int Cout) {
  const int wave = threadIdx.x >> 5;
  const int rbase = blockIdx.x * 32;
  const int obase = blockIdx.y * 256 + wave * 32;
  v8f c00 = {}, c01 = {}, c10 = {}, c11 = {};
  for (int kb = 0; kb < Cin; kb += 32) {
    v16bf a0 = frag_a(X, Cin, rbase, kb);
    v16bf a1 = frag_a(X, Cin, rbase + 16, kb);
    v16bf b0 = frag_b(W, Cin, obase, kb);
    v16bf b1 = frag_b(W, Cin, obase + 16, kb);
    c00 = wmma_bf16(a0, b0, c00);
    c01 = wmma_bf16(a0, b1, c01);
    c10 = wmma_bf16(a1, b0, c10);
    c11 = wmma_bf16(a1, b1, c11);
  }
  store_mish(OUT, Cout, rbase, obase, c00);
  store_mish(OUT, Cout, rbase, obase + 16, c01);
  store_mish(OUT, Cout, rbase + 16, obase, c10);
  store_mish(OUT, Cout, rbase + 16, obase + 16, c11);
}

// 7-tap causal conv as 7 shifted GEMMs, 2x2 blocked:
// OUT[r][o] = mish( sum_k sum_c X[r+k-6][c] * Wk[o][c] )
__global__ void k_conv7_mish(const bf16_t* __restrict__ X, const bf16_t* __restrict__ W,
                             bf16_t* __restrict__ OUT) {
  const int wave = threadIdx.x >> 5;
  const int rbase = blockIdx.x * 32;
  const int obase = blockIdx.y * 256 + wave * 32;
  v8f c00 = {}, c01 = {}, c10 = {}, c11 = {};
  for (int k = 0; k < Kk; ++k) {
    const int shift = k - (Kk - 1);
    const bf16_t* Wk = W + (size_t)k * I3 * Ii;
    for (int kb = 0; kb < Ii; kb += 32) {
      v16bf a0 = frag_a_shift(X, Ii, rbase, shift, kb);
      v16bf a1 = frag_a_shift(X, Ii, rbase + 16, shift, kb);
      v16bf b0 = frag_b(Wk, Ii, obase, kb);
      v16bf b1 = frag_b(Wk, Ii, obase + 16, kb);
      c00 = wmma_bf16(a0, b0, c00);
      c01 = wmma_bf16(a0, b1, c01);
      c10 = wmma_bf16(a1, b0, c10);
      c11 = wmma_bf16(a1, b1, c11);
    }
  }
  store_mish(OUT, I3, rbase, obase, c00);
  store_mish(OUT, I3, rbase, obase + 16, c01);
  store_mish(OUT, I3, rbase + 16, obase, c10);
  store_mish(OUT, I3, rbase + 16, obase + 16, c11);
}

// GEMM + reversible residual, 2x2 blocked: OUT = RES + X @ W^T
__global__ void k_gemm_res(const bf16_t* __restrict__ X, const bf16_t* __restrict__ W,
                           const bf16_t* __restrict__ RES, bf16_t* __restrict__ OUT,
                           int Cin, int Cout) {
  const int wave = threadIdx.x >> 5;
  const int rbase = blockIdx.x * 32;
  const int obase = blockIdx.y * 256 + wave * 32;
  v8f c00 = {}, c01 = {}, c10 = {}, c11 = {};
  for (int kb = 0; kb < Cin; kb += 32) {
    v16bf a0 = frag_a(X, Cin, rbase, kb);
    v16bf a1 = frag_a(X, Cin, rbase + 16, kb);
    v16bf b0 = frag_b(W, Cin, obase, kb);
    v16bf b1 = frag_b(W, Cin, obase + 16, kb);
    c00 = wmma_bf16(a0, b0, c00);
    c01 = wmma_bf16(a0, b1, c01);
    c10 = wmma_bf16(a1, b0, c10);
    c11 = wmma_bf16(a1, b1, c11);
  }
  store_res(OUT, RES, Cout, rbase, obase, c00);
  store_res(OUT, RES, Cout, rbase, obase + 16, c01);
  store_res(OUT, RES, Cout, rbase + 16, obase, c10);
  store_res(OUT, RES, Cout, rbase + 16, obase + 16, c11);
}

// ---------------------------------------------------------------------------
// lin_attn scan (3-phase chunked, f32 accumulation):
//   a[b,s,c] = cumsum_s(depth)/(s+1) * scale + shift
// phase 1: per-chunk depth sums       (B * NCHUNK * I threads)
// phase 2: exclusive prefix of sums   (B * I threads, 32-step serial)
// phase 3: chunk-local scan + combine (B * NCHUNK * I threads)
// ---------------------------------------------------------------------------
__global__ void k_scan_sum(const bf16_t* __restrict__ V, float* __restrict__ P) {
  int t = blockIdx.x * blockDim.x + threadIdx.x;   // (b, chunk, c)
  if (t >= Bb * NCHUNK * Ii) return;
  int c = t % Ii;
  int ch = (t / Ii) % NCHUNK;
  int b = t / (Ii * NCHUNK);
  const bf16_t* vp = V + ((size_t)b * Ss + (size_t)ch * CHLEN) * I3 + c;
  float acc = 0.f;
  for (int i = 0; i < CHLEN; ++i) {
    acc += (float)*vp;
    vp += I3;
  }
  P[((size_t)b * NCHUNK + ch) * Ii + c] = acc;
}

__global__ void k_scan_prefix(float* __restrict__ P) {
  int t = blockIdx.x * blockDim.x + threadIdx.x;   // (b, c)
  if (t >= Bb * Ii) return;
  int c = t % Ii, b = t / Ii;
  float* p = P + (size_t)b * NCHUNK * Ii + c;
  float run = 0.f;
  for (int ch = 0; ch < NCHUNK; ++ch) {
    float v = p[(size_t)ch * Ii];
    p[(size_t)ch * Ii] = run;      // exclusive prefix
    run += v;
  }
}

__global__ void k_scan_apply(const bf16_t* __restrict__ V, const float* __restrict__ P,
                             bf16_t* __restrict__ A) {
  int t = blockIdx.x * blockDim.x + threadIdx.x;   // (b, chunk, c)
  if (t >= Bb * NCHUNK * Ii) return;
  int c = t % Ii;
  int ch = (t / Ii) % NCHUNK;
  int b = t / (Ii * NCHUNK);
  const bf16_t* vp = V + ((size_t)b * Ss + (size_t)ch * CHLEN) * I3 + c;
  bf16_t* ap = A + ((size_t)b * Ss + (size_t)ch * CHLEN) * Ii + c;
  float acc = P[((size_t)b * NCHUNK + ch) * Ii + c];
  int s = ch * CHLEN;
  for (int i = 0; i < CHLEN; ++i) {
    float d  = (float)vp[0];
    float sc = (float)vp[Ii];
    float sh = (float)vp[2 * Ii];
    acc += d;
    *ap = (bf16_t)(acc / (float)(s + 1) * sc + sh);
    vp += I3;
    ap += Ii;
    ++s;
  }
}

// ---------------------------------------------------------------------------
// logits (2x2 blocked): OUT[b][v][s] = x0·ow[v,0:F] + x1·ow[v,F:2F] + bias[v]
// ---------------------------------------------------------------------------
__global__ void k_logits(const bf16_t* __restrict__ X0, const bf16_t* __restrict__ X1,
                         const bf16_t* __restrict__ OW, const float* __restrict__ bias,
                         float* __restrict__ OUT) {
  const int wave = threadIdx.x >> 5;
  const int rbase = blockIdx.x * 32;
  const int vbase = blockIdx.y * 256 + wave * 32;
  v8f c00 = {}, c01 = {}, c10 = {}, c11 = {};
  for (int kb = 0; kb < Ff; kb += 32) {
    v16bf a0 = frag_a(X0, Ff, rbase, kb);
    v16bf a1 = frag_a(X0, Ff, rbase + 16, kb);
    v16bf b0 = frag_b(OW, 2 * Ff, vbase, kb);
    v16bf b1 = frag_b(OW, 2 * Ff, vbase + 16, kb);
    c00 = wmma_bf16(a0, b0, c00);
    c01 = wmma_bf16(a0, b1, c01);
    c10 = wmma_bf16(a1, b0, c10);
    c11 = wmma_bf16(a1, b1, c11);
  }
  for (int kb = 0; kb < Ff; kb += 32) {
    v16bf a0 = frag_a(X1, Ff, rbase, kb);
    v16bf a1 = frag_a(X1, Ff, rbase + 16, kb);
    v16bf b0 = frag_b(OW, 2 * Ff, vbase, Ff + kb);
    v16bf b1 = frag_b(OW, 2 * Ff, vbase + 16, Ff + kb);
    c00 = wmma_bf16(a0, b0, c00);
    c01 = wmma_bf16(a0, b1, c01);
    c10 = wmma_bf16(a1, b0, c10);
    c11 = wmma_bf16(a1, b1, c11);
  }
  store_logits(OUT, bias, rbase, vbase, c00);
  store_logits(OUT, bias, rbase, vbase + 16, c01);
  store_logits(OUT, bias, rbase + 16, vbase, c10);
  store_logits(OUT, bias, rbase + 16, vbase + 16, c11);
}

// ---------------------------------------------------------------------------
// host side
// ---------------------------------------------------------------------------
extern "C" void kernel_launch(void* const* d_in, const int* in_sizes, int n_in,
                              void* d_out, int out_size, void* d_ws, size_t ws_size,
                              hipStream_t stream) {
  const int*   inp  = (const int*)d_in[0];
  const float* emb  = (const float*)d_in[1];
  const float* w0   = (const float*)d_in[2];
  const float* w1   = (const float*)d_in[3];
  const float* w2   = (const float*)d_in[4];
  const float* outw = (const float*)d_in[5];
  const float* outb = (const float*)d_in[6];
  float* OUT = (float*)d_out;

  // bump allocator over workspace (~234 MB total)
  char* wp = (char*)d_ws;
  auto alloc = [&](size_t bytes) -> void* {
    void* r = (void*)wp;
    wp += (bytes + 255) & ~(size_t)255;
    return r;
  };
  bf16_t* x0   = (bf16_t*)alloc((size_t)BSr * Ff * 2);
  bf16_t* x1   = (bf16_t*)alloc((size_t)BSr * Ff * 2);
  bf16_t* xn   = (bf16_t*)alloc((size_t)BSr * Ff * 2);
  bf16_t* vbuf = (bf16_t*)alloc((size_t)BSr * I3 * 2);
  bf16_t* abuf = (bf16_t*)alloc((size_t)BSr * Ii * 2);
  float*  pbuf = (float*)alloc((size_t)Bb * NCHUNK * Ii * 4);
  bf16_t* w0b  = (bf16_t*)alloc((size_t)Dd * I3 * Ff * 2);
  bf16_t* w1b  = (bf16_t*)alloc((size_t)Dd * Kk * I3 * Ii * 2);
  bf16_t* w2b  = (bf16_t*)alloc((size_t)Dd * Ff * Ii * 2);
  bf16_t* owb  = (bf16_t*)alloc((size_t)Vv * 2 * Ff * 2);

  const int T = 256;  // 8 waves of 32

  // weight conversion
  k_cvt_bf16<<<1024, T, 0, stream>>>(w0, w0b, (size_t)Dd * I3 * Ff);
  k_cvt_bf16<<<1024, T, 0, stream>>>(w2, w2b, (size_t)Dd * Ff * Ii);
  k_cvt_bf16<<<512, T, 0, stream>>>(outw, owb, (size_t)Vv * 2 * Ff);
  k_w1_prep<<<4096, T, 0, stream>>>(w1, w1b);

  // embedding
  {
    size_t n = (size_t)BSr * Ff;
    k_embed<<<(unsigned)((n + T - 1) / T), T, 0, stream>>>(inp, emb, x0, x1);
  }

  const int scanN = Bb * NCHUNK * Ii;

  // reversible stack
  for (int d = 0; d < Dd; ++d) {
    const bf16_t* w0d = w0b + (size_t)d * I3 * Ff;
    const bf16_t* w1d = w1b + (size_t)d * Kk * I3 * Ii;
    const bf16_t* w2d = w2b + (size_t)d * Ff * Ii;

    // v = mish(conv1x1(x1, w0))  : [BS][3I]
    k_gemm_mish<<<dim3(BSr / 32, I3 / 256), T, 0, stream>>>(x1, w0d, vbuf, Ff, I3);
    // a = lin_attn(v)            : [BS][I]
    k_scan_sum<<<(scanN + T - 1) / T, T, 0, stream>>>(vbuf, pbuf);
    k_scan_prefix<<<(Bb * Ii + T - 1) / T, T, 0, stream>>>(pbuf);
    k_scan_apply<<<(scanN + T - 1) / T, T, 0, stream>>>(vbuf, pbuf, abuf);
    // v = mish(causal_conv(a, w1))
    k_conv7_mish<<<dim3(BSr / 32, I3 / 256), T, 0, stream>>>(abuf, w1d, vbuf);
    // a = lin_attn(v)
    k_scan_sum<<<(scanN + T - 1) / T, T, 0, stream>>>(vbuf, pbuf);
    k_scan_prefix<<<(Bb * Ii + T - 1) / T, T, 0, stream>>>(pbuf);
    k_scan_apply<<<(scanN + T - 1) / T, T, 0, stream>>>(vbuf, pbuf, abuf);
    // xn = x0 + conv1x1(a, w2)   : [BS][F]
    k_gemm_res<<<dim3(BSr / 32, Ff / 256), T, 0, stream>>>(abuf, w2d, x0, xn, Ii, Ff);

    // (x0, x1) <- (x1, xn)
    bf16_t* old0 = x0;
    x0 = x1;
    x1 = xn;
    xn = old0;
  }

  // final projection -> [B][V][S] fp32
  k_logits<<<dim3(BSr / 32, Vv / 256), T, 0, stream>>>(x0, x1, owb, outb, OUT);
}
